// GeoFNO_32719060861280
// MI455X (gfx1250) — matrix operand order repaired and checked by reading, not compiled, not run on hardware
//
#include <hip/hip_runtime.h>
#include <hip/hip_bf16.h>
#include <math.h>

// ---------------------------------------------------------------------------
// GeoFNO forward on MI455X (gfx1250), bf16 WMMA with fp32 accumulation.
// All big GEMMs use v_wmma_f32_16x16x32_bf16 with 2x2 register blocking per
// wave (8 b128 fragment loads per 8 WMMAs); operands stored with the reduction
// dimension contiguous so each fragment is two global_load_b128 per lane.
// The forward spectral transform is split-K by 4 over the N=8192 reduction
// (deterministic: per-slice partials summed in fixed order by k_mode_mix).
// ---------------------------------------------------------------------------

typedef __bf16 bf16_t;
typedef __attribute__((ext_vector_type(16))) __bf16 v16bf;
typedef __attribute__((ext_vector_type(8)))  __bf16 v8bf;
typedef __attribute__((ext_vector_type(8)))  float  v8f;

#define DEV __device__ __forceinline__

constexpr int B_  = 16;
constexpr int N_  = 8192;
constexpr int K_  = 128;
constexpr int C_  = 128;
constexpr int L_  = 4;
constexpr int FC_ = 128;
constexpr int XSTR = 7;       // IN_DIM + NDIM + 2
constexpr int SPLITK = 4;     // forward-transform reduction slices
constexpr int NSL = N_ / SPLITK;

DEV int laneid() { return (int)(threadIdx.x & 31u); }

// A fragment: matrix stored row-major [M][Kred], ld = row stride (elements).
// ISA 16-bit A 16x32 layout: lane L -> row L%16; elems 0..7 -> K = 8*(L/16)+e,
// elems 8..15 -> K = 16 + 8*(L/16) + (e-8).  Two contiguous 16B loads.
DEV v16bf frag_a(const bf16_t* p, int ld) {
  int l = laneid(); int m = l & 15; int h = l >> 4;
  const bf16_t* r = p + (size_t)m * (size_t)ld;
  v8bf lo = *(const v8bf*)(r + h * 8);
  v8bf hi = *(const v8bf*)(r + 16 + h * 8);
  return __builtin_shufflevector(lo, hi, 0,1,2,3,4,5,6,7,8,9,10,11,12,13,14,15);
}

// B fragment: B^T stored row-major [Ncol][Kred] (reduction contiguous).
// ISA 16-bit B 32x16 layout: lane L -> col L%16; elems 0..7 -> K = 8*(L/16)+e,
// elems 8..15 -> K = 16 + 8*(L/16) + (e-8).
DEV v16bf frag_b(const bf16_t* p, int ld) {
  int l = laneid(); int nc = l & 15; int h = l >> 4;
  const bf16_t* r = p + (size_t)nc * (size_t)ld;
  v8bf lo = *(const v8bf*)(r + h * 8);
  v8bf hi = *(const v8bf*)(r + 16 + h * 8);
  return __builtin_shufflevector(lo, hi, 0,1,2,3,4,5,6,7,8,9,10,11,12,13,14,15);
}

DEV void wmma_acc(v8f& acc, v16bf a, v16bf b) {
  acc = __builtin_amdgcn_wmma_f32_16x16x32_bf16(
      /*neg_a=*/false, a, /*neg_b=*/false, b,
      /*c_mod=*/(short)0, acc, /*reuse_a=*/false, /*reuse_b=*/false);
}

DEV float gelu_exact(float v) { return 0.5f * v * (1.0f + erff(v * 0.70710678118654752f)); }

// ---------------------------------------------------------------------------
// Precompute bases once (shared by all 4 layers).
//   wb_c[b][k][n] =  cos(g.m_k)*mask*w*N      (forward-transform B operand)
//   wb_s[b][k][n] = -sin(g.m_k)*mask*w*N      (minus folded: x_s = -<h,wbases_s>)
//   b_cT[b][n][k] =  cos*mask,  b_sT = sin*mask  (inverse-transform B operand)
// ---------------------------------------------------------------------------
__global__ void k_precompute(const float* __restrict__ x, const float* __restrict__ modes,
                             bf16_t* __restrict__ wb_c, bf16_t* __restrict__ wb_s,
                             bf16_t* __restrict__ b_cT, bf16_t* __restrict__ b_sT,
                             float* __restrict__ wsz0, float* __restrict__ maskv) {
  __shared__ float sm[K_ * 2];
  for (int i = threadIdx.x; i < K_ * 2; i += blockDim.x) sm[i] = modes[i];
  __syncthreads();
  int t = blockIdx.x * blockDim.x + threadIdx.x;   // 0 .. B*N-1
  int b = t / N_, n = t % N_;
  const float* xr = x + (size_t)t * XSTR;
  float gx = xr[3], gy = xr[4], w = xr[5], mk = xr[6];
  float wsz = w * (float)N_;
  wsz0[t]  = mk * wsz;
  maskv[t] = mk;
  for (int k = 0; k < K_; ++k) {
    float th  = gx * sm[2 * k] + gy * sm[2 * k + 1];
    float cth = cosf(th) * mk;
    float sth = sinf(th) * mk;
    wb_c[((size_t)b * K_ + k) * N_ + n] = (bf16_t)(cth * wsz);
    wb_s[((size_t)b * K_ + k) * N_ + n] = (bf16_t)(-sth * wsz);
    b_cT[(size_t)t * K_ + k] = (bf16_t)cth;
    b_sT[(size_t)t * K_ + k] = (bf16_t)sth;
  }
}

// h[b][c][n] = sum_i x[b][n][i] fc0_w[i][c] + fc0_b[c]; dual bf16 layouts.
__global__ void k_fc0(const float* __restrict__ x, const float* __restrict__ fc0_w,
                      const float* __restrict__ fc0_b,
                      bf16_t* __restrict__ hbf, bf16_t* __restrict__ hTbf) {
  int t = blockIdx.x * blockDim.x + threadIdx.x;   // b*N+n
  int b = t / N_, n = t % N_;
  const float* xr = x + (size_t)t * XSTR;
  float i0 = xr[0], i1 = xr[1], i2 = xr[2];
  for (int c = 0; c < C_; ++c) {
    float v = i0 * fc0_w[c] + i1 * fc0_w[C_ + c] + i2 * fc0_w[2 * C_ + c] + fc0_b[c];
    hbf[((size_t)b * C_ + c) * N_ + n] = (bf16_t)v;
    hTbf[(size_t)t * C_ + c]           = (bf16_t)v;
  }
}

__global__ void k_cvt_convw(const float* __restrict__ conv_w, bf16_t* __restrict__ cw) {
  int t = blockIdx.x * blockDim.x + threadIdx.x;
  if (t < L_ * C_ * C_) cw[t] = (bf16_t)conv_w[t];
}

__global__ void k_cvt_fc1T(const float* __restrict__ fc1_w, bf16_t* __restrict__ fc1T) {
  int t = blockIdx.x * blockDim.x + threadIdx.x;   // c*FC+f
  if (t < C_ * FC_) {
    int c = t / FC_, f = t % FC_;
    fc1T[(size_t)f * C_ + c] = (bf16_t)fc1_w[t];
  }
}

// ---------------------------------------------------------------------------
// Forward transform, split-K: x_c_part[s][b][i][k] = sum_{n in slice s}
// h[i][n] wb_c[k][n]  (and x_s).  2x2 register blocking: each wave owns a
// 32(c) x 32(k) super-tile and both cos/sin accumulators (8 x v8f); per 32-n
// step: 6 fragment loads, 8 WMMAs.  1024 waves total.
// ---------------------------------------------------------------------------
__global__ void k_spectral_fwd(const bf16_t* __restrict__ hbf,
                               const bf16_t* __restrict__ wb_c,
                               const bf16_t* __restrict__ wb_s,
                               float* __restrict__ x_c_part, float* __restrict__ x_s_part) {
  int wave = threadIdx.x >> 5;
  int tb    = blockIdx.x & 1;        // which half of the 16 super-tiles
  int slice = blockIdx.x >> 1;       // 0..SPLITK-1 reduction slice
  int id = tb * 8 + wave;            // 0..15 super-tiles
  int cs = id >> 2, ks = id & 3;     // 32-row / 32-col super-tile coords
  int b = blockIdx.y;
  int n0 = slice * NSL;
  const bf16_t* ha0 = hbf  + ((size_t)b * C_ + cs * 32)      * N_ + n0;
  const bf16_t* ha1 = hbf  + ((size_t)b * C_ + cs * 32 + 16) * N_ + n0;
  const bf16_t* wc0 = wb_c + ((size_t)b * K_ + ks * 32)      * N_ + n0;
  const bf16_t* wc1 = wb_c + ((size_t)b * K_ + ks * 32 + 16) * N_ + n0;
  const bf16_t* ws0 = wb_s + ((size_t)b * K_ + ks * 32)      * N_ + n0;
  const bf16_t* ws1 = wb_s + ((size_t)b * K_ + ks * 32 + 16) * N_ + n0;
  v8f ac[2][2] = {}; v8f as[2][2] = {};
  for (int nn = 0; nn < NSL; nn += 32) {
    // Speculative stream-ahead into GL2 (global_prefetch_b8); no guard branch.
    __builtin_prefetch(ha0 + nn + 256, 0, 0);
    __builtin_prefetch(ha1 + nn + 256, 0, 0);
    v16bf a0  = frag_a(ha0 + nn, N_);
    v16bf a1  = frag_a(ha1 + nn, N_);
    v16bf bc0 = frag_b(wc0 + nn, N_);
    v16bf bc1 = frag_b(wc1 + nn, N_);
    v16bf bs0 = frag_b(ws0 + nn, N_);
    v16bf bs1 = frag_b(ws1 + nn, N_);
    wmma_acc(ac[0][0], a0, bc0); wmma_acc(ac[0][1], a0, bc1);
    wmma_acc(ac[1][0], a1, bc0); wmma_acc(ac[1][1], a1, bc1);
    wmma_acc(as[0][0], a0, bs0); wmma_acc(as[0][1], a0, bs1);
    wmma_acc(as[1][0], a1, bs0); wmma_acc(as[1][1], a1, bs1);
  }
  const size_t sl_off = (size_t)slice * B_ * C_ * K_;
  int l = laneid(); int col = l & 15; int h = l >> 4;
  for (int u = 0; u < 2; ++u)
    for (int w = 0; w < 2; ++w)
      for (int v = 0; v < 8; ++v) {
        int row  = cs * 32 + u * 16 + v + 8 * h;
        int kcol = ks * 32 + w * 16 + col;
        x_c_part[sl_off + ((size_t)b * C_ + row) * K_ + kcol] = ac[u][w][v];
        x_s_part[sl_off + ((size_t)b * C_ + row) * K_ + kcol] = as[u][w][v];
      }
}

// x_0[b][i] = sum_n h[b][i][n] * (mask*w*N)[b][n]
__global__ void k_reduce_x0(const bf16_t* __restrict__ hbf, const float* __restrict__ wsz0,
                            float* __restrict__ x_0) {
  int i = blockIdx.x, b = blockIdx.y;
  __shared__ float red[128];
  const bf16_t* hr = hbf + ((size_t)b * C_ + i) * N_;
  const float*  wr = wsz0 + (size_t)b * N_;
  float s = 0.f;
  for (int n = threadIdx.x; n < N_; n += 128) s += (float)hr[n] * wr[n];
  red[threadIdx.x] = s; __syncthreads();
  for (int st = 64; st > 0; st >>= 1) {
    if ((int)threadIdx.x < st) red[threadIdx.x] += red[threadIdx.x + st];
    __syncthreads();
  }
  if (threadIdx.x == 0) x_0[(size_t)b * C_ + i] = red[0];
}

// Per-mode mixing (reduction over C=128 per (b,o,k)).  Sums the SPLITK
// forward-transform partials in fixed order (deterministic), and folds the
// 1/size, the +/-2 of the reconstruction, and the x_s/x_c cross terms.
__global__ void k_mode_mix(const float* __restrict__ x_c_part, const float* __restrict__ x_s_part,
                           const float* __restrict__ Wc, const float* __restrict__ Ws,
                           bf16_t* __restrict__ fcbf, bf16_t* __restrict__ fsbf) {
  int t = blockIdx.x * blockDim.x + threadIdx.x;   // (b*C+o)*K+k
  int k = t % K_; int o = (t / K_) % C_; int b = t / (C_ * K_);
  const size_t SL = (size_t)B_ * C_ * K_;
  float fc = 0.f, fs = 0.f;
  for (int i = 0; i < C_; ++i) {
    size_t e = ((size_t)b * C_ + i) * K_ + k;
    float xc = (x_c_part[e] + x_c_part[e + SL]) + (x_c_part[e + 2 * SL] + x_c_part[e + 3 * SL]);
    float xs = (x_s_part[e] + x_s_part[e + SL]) + (x_s_part[e + 2 * SL] + x_s_part[e + 3 * SL]);
    float wc = Wc[((size_t)i * C_ + o) * K_ + k];
    float ws = Ws[((size_t)i * C_ + o) * K_ + k];
    fc += xc * wc - xs * ws;
    fs += xs * wc + xc * ws;
  }
  const float inv = 1.0f / (float)N_;
  fcbf[t] = (bf16_t)( 2.0f * fc * inv);   // +2*f_c
  fsbf[t] = (bf16_t)(-2.0f * fs * inv);   // -2*f_s
}

__global__ void k_mix_f0(const float* __restrict__ x_0, const float* __restrict__ w0l,
                         float* __restrict__ f_0) {
  int t = blockIdx.x * blockDim.x + threadIdx.x;   // b*C+o
  int o = t % C_; int b = t / C_;
  float s = 0.f;
  for (int i = 0; i < C_; ++i) s += x_0[(size_t)b * C_ + i] * w0l[(size_t)i * C_ + o];
  f_0[t] = s / (float)N_;
}

// ---------------------------------------------------------------------------
// Fused inverse transform + 1x1 conv + bias + f_0*mask (+ GELU).
// 2x2 register blocking: wave owns 32(o) x 32(n); double-buffered h.
// ---------------------------------------------------------------------------
__global__ void k_inverse_conv(const bf16_t* __restrict__ fcbf, const bf16_t* __restrict__ fsbf,
                               const bf16_t* __restrict__ b_cT, const bf16_t* __restrict__ b_sT,
                               const bf16_t* __restrict__ convw, const bf16_t* __restrict__ hT_in,
                               const float* __restrict__ f_0, const float* __restrict__ maskv,
                               const float* __restrict__ convb, int apply_gelu,
                               bf16_t* __restrict__ hbf_out, bf16_t* __restrict__ hTbf_out) {
  int wave = threadIdx.x >> 5;
  int id = blockIdx.x * 8 + wave;   // 0..1023 super-tiles
  int os = id >> 8, ns = id & 255;  // 4 x 256
  int b = blockIdx.y;
  v8f acc[2][2] = {};
  const bf16_t* fa0 = fcbf + ((size_t)b * C_ + os * 32)      * K_;
  const bf16_t* fa1 = fcbf + ((size_t)b * C_ + os * 32 + 16) * K_;
  const bf16_t* fb0 = fsbf + ((size_t)b * C_ + os * 32)      * K_;
  const bf16_t* fb1 = fsbf + ((size_t)b * C_ + os * 32 + 16) * K_;
  const bf16_t* bc0 = b_cT + ((size_t)b * N_ + ns * 32)      * K_;
  const bf16_t* bc1 = b_cT + ((size_t)b * N_ + ns * 32 + 16) * K_;
  const bf16_t* bs0 = b_sT + ((size_t)b * N_ + ns * 32)      * K_;
  const bf16_t* bs1 = b_sT + ((size_t)b * N_ + ns * 32 + 16) * K_;
  for (int kk = 0; kk < K_; kk += 32) {
    v16bf a0 = frag_a(fa0 + kk, K_), a1 = frag_a(fa1 + kk, K_);
    v16bf c0 = frag_b(bc0 + kk, K_), c1 = frag_b(bc1 + kk, K_);
    wmma_acc(acc[0][0], a0, c0); wmma_acc(acc[0][1], a0, c1);
    wmma_acc(acc[1][0], a1, c0); wmma_acc(acc[1][1], a1, c1);
    v16bf d0 = frag_a(fb0 + kk, K_), d1 = frag_a(fb1 + kk, K_);
    v16bf s0 = frag_b(bs0 + kk, K_), s1 = frag_b(bs1 + kk, K_);
    wmma_acc(acc[0][0], d0, s0); wmma_acc(acc[0][1], d0, s1);
    wmma_acc(acc[1][0], d1, s0); wmma_acc(acc[1][1], d1, s1);
  }
  const bf16_t* ca0 = convw + (size_t)(os * 32)      * C_;
  const bf16_t* ca1 = convw + (size_t)(os * 32 + 16) * C_;
  const bf16_t* hb0 = hT_in + ((size_t)b * N_ + ns * 32)      * C_;
  const bf16_t* hb1 = hT_in + ((size_t)b * N_ + ns * 32 + 16) * C_;
  for (int ii = 0; ii < C_; ii += 32) {
    v16bf a0 = frag_a(ca0 + ii, C_), a1 = frag_a(ca1 + ii, C_);
    v16bf h0 = frag_b(hb0 + ii, C_), h1 = frag_b(hb1 + ii, C_);
    wmma_acc(acc[0][0], a0, h0); wmma_acc(acc[0][1], a0, h1);
    wmma_acc(acc[1][0], a1, h0); wmma_acc(acc[1][1], a1, h1);
  }
  int l = laneid(); int col = l & 15; int h = l >> 4;
  for (int un = 0; un < 2; ++un) {
    int n = ns * 32 + un * 16 + col;
    float mk = maskv[(size_t)b * N_ + n];
    for (int uo = 0; uo < 2; ++uo)
      for (int v = 0; v < 8; ++v) {
        int o = os * 32 + uo * 16 + v + 8 * h;
        float val = acc[uo][un][v] + f_0[(size_t)b * C_ + o] * mk + convb[o];
        if (apply_gelu) val = gelu_exact(val);
        hbf_out[((size_t)b * C_ + o) * N_ + n] = (bf16_t)val;
        hTbf_out[((size_t)b * N_ + n) * C_ + o] = (bf16_t)val;
      }
  }
}

// y1[b][n][f] = gelu(sum_c hT[n][c] fc1_wT[f][c] + fc1_b[f]); 2x2 blocked.
__global__ void k_mlp1(const bf16_t* __restrict__ hT, const bf16_t* __restrict__ fc1T,
                       const float* __restrict__ fc1_b, float* __restrict__ y1) {
  int wave = threadIdx.x >> 5;
  int id = blockIdx.x * 8 + wave;   // 0..1023 super-tiles
  int fs = id >> 8, ns = id & 255;  // 4 x 256
  int b = blockIdx.y;
  v8f acc[2][2] = {};
  const bf16_t* A0 = hT   + ((size_t)b * N_ + ns * 32)      * C_;
  const bf16_t* A1 = hT   + ((size_t)b * N_ + ns * 32 + 16) * C_;
  const bf16_t* B0 = fc1T + (size_t)(fs * 32)      * C_;
  const bf16_t* B1 = fc1T + (size_t)(fs * 32 + 16) * C_;
  for (int cc = 0; cc < C_; cc += 32) {
    v16bf a0 = frag_a(A0 + cc, C_), a1 = frag_a(A1 + cc, C_);
    v16bf b0 = frag_b(B0 + cc, C_), b1 = frag_b(B1 + cc, C_);
    wmma_acc(acc[0][0], a0, b0); wmma_acc(acc[0][1], a0, b1);
    wmma_acc(acc[1][0], a1, b0); wmma_acc(acc[1][1], a1, b1);
  }
  int l = laneid(); int col = l & 15; int h = l >> 4;
  for (int uf = 0; uf < 2; ++uf) {
    int f = fs * 32 + uf * 16 + col;
    float bias = fc1_b[f];
    for (int un = 0; un < 2; ++un)
      for (int v = 0; v < 8; ++v) {
        int n = ns * 32 + un * 16 + v + 8 * h;
        y1[((size_t)b * N_ + n) * FC_ + f] = gelu_exact(acc[un][uf][v] + bias);
      }
  }
}

__global__ void k_mlp2(const float* __restrict__ y1, const float* __restrict__ fc2_w,
                       const float* __restrict__ fc2_b, float* __restrict__ out) {
  int t = blockIdx.x * blockDim.x + threadIdx.x;   // b*N+n
  const float* r = y1 + (size_t)t * FC_;
  float s = fc2_b[0];
  for (int f = 0; f < FC_; ++f) s += r[f] * fc2_w[f];
  out[t] = s;
}

// ---------------------------------------------------------------------------
extern "C" void kernel_launch(void* const* d_in, const int* in_sizes, int n_in,
                              void* d_out, int out_size, void* d_ws, size_t ws_size,
                              hipStream_t stream) {
  (void)in_sizes; (void)n_in; (void)out_size; (void)ws_size;
  const float* x      = (const float*)d_in[0];
  const float* modes  = (const float*)d_in[1];
  const float* fc0_w  = (const float*)d_in[2];
  const float* fc0_b  = (const float*)d_in[3];
  const float* wc     = (const float*)d_in[4];
  const float* wss    = (const float*)d_in[5];
  const float* w0     = (const float*)d_in[6];
  const float* conv_w = (const float*)d_in[7];
  const float* conv_b = (const float*)d_in[8];
  const float* fc1_w  = (const float*)d_in[9];
  const float* fc1_b  = (const float*)d_in[10];
  const float* fc2_w  = (const float*)d_in[11];
  const float* fc2_b  = (const float*)d_in[12];
  float* out = (float*)d_out;

  // Workspace carve-out (~346 MB total).
  char* p = (char*)d_ws;
  auto take = [&](size_t bytes) -> char* {
    char* r = p; p += (bytes + 255) & ~(size_t)255; return r;
  };
  bf16_t* wb_c   = (bf16_t*)take((size_t)B_ * K_ * N_ * 2);
  bf16_t* wb_s   = (bf16_t*)take((size_t)B_ * K_ * N_ * 2);
  bf16_t* b_cT   = (bf16_t*)take((size_t)B_ * N_ * K_ * 2);
  bf16_t* b_sT   = (bf16_t*)take((size_t)B_ * N_ * K_ * 2);
  float*  wsz0   = (float*) take((size_t)B_ * N_ * 4);
  float*  maskv  = (float*) take((size_t)B_ * N_ * 4);
  bf16_t* hbf[2]  = { (bf16_t*)take((size_t)B_ * C_ * N_ * 2),
                      (bf16_t*)take((size_t)B_ * C_ * N_ * 2) };
  bf16_t* hTbf[2] = { (bf16_t*)take((size_t)B_ * N_ * C_ * 2),
                      (bf16_t*)take((size_t)B_ * N_ * C_ * 2) };
  float*  x_c    = (float*) take((size_t)SPLITK * B_ * C_ * K_ * 4);
  float*  x_s    = (float*) take((size_t)SPLITK * B_ * C_ * K_ * 4);
  float*  x_0    = (float*) take((size_t)B_ * C_ * 4);
  bf16_t* fcbf   = (bf16_t*)take((size_t)B_ * C_ * K_ * 2);
  bf16_t* fsbf   = (bf16_t*)take((size_t)B_ * C_ * K_ * 2);
  float*  f_0    = (float*) take((size_t)B_ * C_ * 4);
  bf16_t* convwb = (bf16_t*)take((size_t)L_ * C_ * C_ * 2);
  bf16_t* fc1T   = (bf16_t*)take((size_t)C_ * FC_ * 2);
  float*  y1     = (float*) take((size_t)B_ * N_ * FC_ * 4);

  // Stage 0: bases, fc0, weight conversions.
  k_precompute<<<(B_ * N_) / 256, 256, 0, stream>>>(x, modes, wb_c, wb_s, b_cT, b_sT, wsz0, maskv);
  k_fc0<<<(B_ * N_) / 256, 256, 0, stream>>>(x, fc0_w, fc0_b, hbf[0], hTbf[0]);
  k_cvt_convw<<<(L_ * C_ * C_) / 256, 256, 0, stream>>>(conv_w, convwb);
  k_cvt_fc1T<<<(C_ * FC_) / 256, 256, 0, stream>>>(fc1_w, fc1T);

  // Spectral layers, double-buffered h.
  int cur = 0;
  for (int l = 0; l < L_; ++l) {
    int nxt = cur ^ 1;
    k_spectral_fwd<<<dim3(2 * SPLITK, B_), 256, 0, stream>>>(hbf[cur], wb_c, wb_s, x_c, x_s);
    k_reduce_x0<<<dim3(C_, B_), 128, 0, stream>>>(hbf[cur], wsz0, x_0);
    k_mode_mix<<<(B_ * C_ * K_) / 256, 256, 0, stream>>>(
        x_c, x_s, wc + (size_t)l * C_ * C_ * K_, wss + (size_t)l * C_ * C_ * K_, fcbf, fsbf);
    k_mix_f0<<<(B_ * C_) / 256, 256, 0, stream>>>(x_0, w0 + (size_t)l * C_ * C_, f_0);
    k_inverse_conv<<<dim3(128, B_), 256, 0, stream>>>(
        fcbf, fsbf, b_cT, b_sT, convwb + (size_t)l * C_ * C_, hTbf[cur],
        f_0, maskv, conv_b + (size_t)l * C_, (l < L_ - 1) ? 1 : 0, hbf[nxt], hTbf[nxt]);
    cur = nxt;
  }

  // Final MLP.
  k_mlp1<<<dim3(128, B_), 256, 0, stream>>>(hTbf[cur], fc1T, fc1_b, y1);
  k_mlp2<<<(B_ * N_) / 256, 256, 0, stream>>>(y1, fc2_w, fc2_b, out);
}
